// AttentionLSTM_47854525612159
// MI455X (gfx1250) — compile-verified, matrix operand order checked by reading, not run on hardware
//
#include <hip/hip_runtime.h>
#include <hip/hip_bf16.h>

// ---------------------------------------------------------------------------
// AttentionLSTM for MI455X (gfx1250, wave32, WMMA)
//   N=512 batch, T=64 steps, D=512 in, H=512 hidden, 4H=2048
//
// Per step:  a = [x_t | h | attn] (512 x 1536, bf16)  @  Wcat^T (1536 x 2048)
// computed with v_wmma_f32_16x16x32_bf16, fp32 accumulate. Gates/softmax fp32.
// Each wave owns a 32(M) x 64(N) output tile: 2 A frags x 4 B frags = 8 WMMAs
// per K-chunk of 32, K split into three branch-free region loops.
// ---------------------------------------------------------------------------

typedef __bf16 bf16_t;
typedef __attribute__((ext_vector_type(16))) __bf16 v16bf;
typedef __attribute__((ext_vector_type(8)))  __bf16 v8bf;
typedef __attribute__((ext_vector_type(8)))  float  v8f;

#define N_BATCH 512
#define T_STEPS 64
#define D_IN    512
#define H_DIM   512
#define FOURH   2048
#define KTOT    1536          // D + H + H (concatenated GEMM K)

union Frag16 { v16bf v; v8bf h[2]; };

// ---------------- prep kernels ----------------

__global__ void k_cast_bf16(const float* __restrict__ src, bf16_t* __restrict__ dst) {
    int i = blockIdx.x * 256 + threadIdx.x;           // exact grid, no bounds check
    dst[i] = (bf16_t)src[i];
}

// Build Wcat^T (4H x KTOT) bf16, row n = [Wx[:,n] | Wh[:,n] | Wattn[:,n]]
__global__ void k_build_wt(const float* __restrict__ Wx, const float* __restrict__ Wh,
                           const float* __restrict__ Wattn, bf16_t* __restrict__ WT) {
    int i = blockIdx.x * 256 + threadIdx.x;           // 2048*1536 total (exact)
    int n = i / KTOT;
    int k = i - n * KTOT;
    float v;
    if (k < D_IN)               v = Wx[(size_t)k * FOURH + n];
    else if (k < D_IN + H_DIM)  v = Wh[(size_t)(k - D_IN) * FOURH + n];
    else                        v = Wattn[(size_t)(k - D_IN - H_DIM) * FOURH + n];
    WT[i] = (bf16_t)v;
}

// h0 = c0 = mean over 16 of A_flat(N,H,16)
__global__ void k_init(const float* __restrict__ A, float* __restrict__ hf,
                       bf16_t* __restrict__ hb, float* __restrict__ c) {
    int i = blockIdx.x * 256 + threadIdx.x;           // N*H (exact)
    const float* a = A + (size_t)i * 16;
    float s = 0.f;
#pragma unroll
    for (int p = 0; p < 16; ++p) s += a[p];
    s *= (1.0f / 16.0f);
    hf[i] = s;
    hb[i] = (bf16_t)s;
    c[i]  = s;
}

// ---------------- attention (per step): one block per batch row ----------------

__global__ void k_attn(const float* __restrict__ A, const float* __restrict__ hf,
                       bf16_t* __restrict__ attnb) {
    const int n   = blockIdx.x;
    const int tid = threadIdx.x;                      // 256 threads, 2 h-rows each
    __shared__ float sc[16];
    __shared__ float w[16];
    if (tid < 16) sc[tid] = 0.f;
    __syncthreads();

    const float* Af = A + (size_t)n * H_DIM * 16;
    float loc[16];
#pragma unroll
    for (int p = 0; p < 16; ++p) loc[p] = 0.f;

#pragma unroll
    for (int r = 0; r < 2; ++r) {
        int h = tid + r * 256;
        float hv = hf[(size_t)n * H_DIM + h];
        const float* row = Af + (size_t)h * 16;
#pragma unroll
        for (int p = 0; p < 16; ++p) loc[p] += hv * row[p];
    }
#pragma unroll
    for (int p = 0; p < 16; ++p) atomicAdd(&sc[p], loc[p]);
    __syncthreads();

    if (tid == 0) {
        const float scale = 0.04419417382415922f;     // 1/sqrt(512)
        float mx = -1e30f;
#pragma unroll
        for (int p = 0; p < 16; ++p) mx = fmaxf(mx, sc[p] * scale);
        float sum = 0.f;
#pragma unroll
        for (int p = 0; p < 16; ++p) { float e = __expf(sc[p] * scale - mx); w[p] = e; sum += e; }
        float inv = 1.f / sum;
#pragma unroll
        for (int p = 0; p < 16; ++p) w[p] *= inv;
    }
    __syncthreads();

#pragma unroll
    for (int r = 0; r < 2; ++r) {
        int h = tid + r * 256;
        const float* row = Af + (size_t)h * 16;
        float acc = 0.f;
#pragma unroll
        for (int p = 0; p < 16; ++p) acc += row[p] * w[p];
        attnb[(size_t)n * H_DIM + h] = (bf16_t)acc;
    }
}

// ---------------- fused recurrent GEMM (per step) ----------------

__device__ __forceinline__ v8f wmma_bf16(const Frag16& a, const Frag16& b, v8f c) {
    return __builtin_amdgcn_wmma_f32_16x16x32_bf16(false, a.v, false, b.v,
                                                   (short)0, c, false, false);
}

// One branch-free K region of 512 (16 chunks of 32).
// ar0/ar1: A row pointers (region-local) for the two M sub-tiles.
// br0..3:  B row pointers; region selected via kbase offset into the WT row.
__device__ __forceinline__ void mma_region(
    const bf16_t* __restrict__ ar0, const bf16_t* __restrict__ ar1,
    const bf16_t* __restrict__ br0, const bf16_t* __restrict__ br1,
    const bf16_t* __restrict__ br2, const bf16_t* __restrict__ br3,
    int kbase, int hi, v8f acc[8]) {
    for (int k0 = 0; k0 < 512; k0 += 32) {
        // lanes 0-15: K = {k0..k0+7, k0+16..k0+23}; lanes 16-31: +8 (ISA striping)
        const int ka = k0 + hi * 8;
        const int kb = kbase + ka;
        Frag16 a0, a1;
        a0.h[0] = *(const v8bf*)(ar0 + ka);  a0.h[1] = *(const v8bf*)(ar0 + ka + 16);
        a1.h[0] = *(const v8bf*)(ar1 + ka);  a1.h[1] = *(const v8bf*)(ar1 + ka + 16);

        Frag16 b;
        b.h[0] = *(const v8bf*)(br0 + kb);   b.h[1] = *(const v8bf*)(br0 + kb + 16);
        acc[0] = wmma_bf16(a0, b, acc[0]);
        acc[1] = wmma_bf16(a1, b, acc[1]);

        b.h[0] = *(const v8bf*)(br1 + kb);   b.h[1] = *(const v8bf*)(br1 + kb + 16);
        acc[2] = wmma_bf16(a0, b, acc[2]);
        acc[3] = wmma_bf16(a1, b, acc[3]);

        b.h[0] = *(const v8bf*)(br2 + kb);   b.h[1] = *(const v8bf*)(br2 + kb + 16);
        acc[4] = wmma_bf16(a0, b, acc[4]);
        acc[5] = wmma_bf16(a1, b, acc[5]);

        b.h[0] = *(const v8bf*)(br3 + kb);   b.h[1] = *(const v8bf*)(br3 + kb + 16);
        acc[6] = wmma_bf16(a0, b, acc[6]);
        acc[7] = wmma_bf16(a1, b, acc[7]);
    }
}

// grid(4, 16), block 256 = 8 waves. Wave w: mBase = blockIdx.y*32,
// nBase = blockIdx.x*512 + w*64 -> 32(M) x 64(N) output tile, 8 accumulators.
__global__ __launch_bounds__(256)
void k_gemm(const bf16_t* __restrict__ xbf, const bf16_t* __restrict__ hb,
            const bf16_t* __restrict__ attnb, const bf16_t* __restrict__ WT,
            float* __restrict__ aout, int t) {
    const int wave = threadIdx.x >> 5;
    const int lane = threadIdx.x & 31;
    const int lr   = lane & 15;                        // index within 16
    const int hi   = lane >> 4;                        // K-half select (ISA striping)
    const int mBase = blockIdx.y * 32;
    const int nBase = blockIdx.x * 512 + wave * 64;
    const int m0 = mBase + lr;                         // A sub-tile 0 row
    const int m1 = m0 + 16;                            // A sub-tile 1 row

    // A-matrix row pointers for the three K regions (row-major bf16 activations)
    const bf16_t* x0 = xbf   + ((size_t)m0 * T_STEPS + t) * D_IN;
    const bf16_t* x1 = xbf   + ((size_t)m1 * T_STEPS + t) * D_IN;
    const bf16_t* h0 = hb    + (size_t)m0 * H_DIM;
    const bf16_t* h1 = hb    + (size_t)m1 * H_DIM;
    const bf16_t* a0 = attnb + (size_t)m0 * H_DIM;
    const bf16_t* a1 = attnb + (size_t)m1 * H_DIM;

    // B-matrix row pointers (WT is 4H x KTOT row-major -> contiguous per lane)
    const bf16_t* b0 = WT + (size_t)(nBase +  0 + lr) * KTOT;
    const bf16_t* b1 = WT + (size_t)(nBase + 16 + lr) * KTOT;
    const bf16_t* b2 = WT + (size_t)(nBase + 32 + lr) * KTOT;
    const bf16_t* b3 = WT + (size_t)(nBase + 48 + lr) * KTOT;

    v8f acc[8] = {};

    mma_region(x0, x1, b0, b1, b2, b3,    0, hi, acc);   // K 0..511    : x_t @ Wx
    mma_region(h0, h1, b0, b1, b2, b3,  512, hi, acc);   // K 512..1023 : h   @ Wh
    mma_region(a0, a1, b0, b1, b2, b3, 1024, hi, acc);   // K 1024..1535: attn@ Wattn

    // C/D layout: VGPR r, lanes 0-15 -> (m = mBase+r,   n = lane)
    //                     lanes 16-31 -> (m = mBase+8+r, n = lane-16)
    const int mr0 = mBase + hi * 8;
    const int mr1 = mr0 + 16;
#pragma unroll
    for (int r = 0; r < 8; ++r) {
        aout[(size_t)(mr0 + r) * FOURH + (nBase +  0 + lr)] = acc[0][r];
        aout[(size_t)(mr1 + r) * FOURH + (nBase +  0 + lr)] = acc[1][r];
        aout[(size_t)(mr0 + r) * FOURH + (nBase + 16 + lr)] = acc[2][r];
        aout[(size_t)(mr1 + r) * FOURH + (nBase + 16 + lr)] = acc[3][r];
        aout[(size_t)(mr0 + r) * FOURH + (nBase + 32 + lr)] = acc[4][r];
        aout[(size_t)(mr1 + r) * FOURH + (nBase + 32 + lr)] = acc[5][r];
        aout[(size_t)(mr0 + r) * FOURH + (nBase + 48 + lr)] = acc[6][r];
        aout[(size_t)(mr1 + r) * FOURH + (nBase + 48 + lr)] = acc[7][r];
    }
}

// ---------------- LSTM gates / state update (per step) ----------------

__global__ void k_gate(const float* __restrict__ aout, const float* __restrict__ b,
                       float* __restrict__ c, float* __restrict__ hf,
                       bf16_t* __restrict__ hb, float* __restrict__ out, int t) {
    int i = blockIdx.x * 256 + threadIdx.x;           // N*H (exact)
    int n = i >> 9;
    int h = i & 511;
    const float* ar = aout + (size_t)n * FOURH;
    float ai = ar[h]            + b[h];
    float af = ar[ 512 + h]     + b[ 512 + h];
    float ao = ar[1024 + h]     + b[1024 + h];
    float ag = ar[1536 + h]     + b[1536 + h];
    float ig = 1.f / (1.f + __expf(-ai));
    float fg = 1.f / (1.f + __expf(-af));
    float og = 1.f / (1.f + __expf(-ao));
    float gg = tanhf(ag);
    float cn = fg * c[i] + ig * gg;
    float hn = og * tanhf(cn);
    c[i]  = cn;
    hf[i] = hn;
    hb[i] = (bf16_t)hn;
    out[((size_t)n * T_STEPS + t) * H_DIM + h] = hn;
}

// ---------------- host orchestration ----------------

extern "C" void kernel_launch(void* const* d_in, const int* in_sizes, int n_in,
                              void* d_out, int out_size, void* d_ws, size_t ws_size,
                              hipStream_t stream) {
    const float* x     = (const float*)d_in[0];   // (N,T,D)
    const float* A     = (const float*)d_in[1];   // (N,H,16)
    const float* Wx    = (const float*)d_in[2];   // (D,4H)
    const float* Wh    = (const float*)d_in[3];   // (H,4H)
    const float* Wattn = (const float*)d_in[4];   // (H,4H)
    const float* b     = (const float*)d_in[5];   // (4H)
    float* out = (float*)d_out;                   // (N,T,H)

    // workspace layout (all offsets 256B-aligned)
    char* base = (char*)d_ws;
    size_t off = 0;
    bf16_t* xbf   = (bf16_t*)(base + off); off += (size_t)N_BATCH * T_STEPS * D_IN * 2;  // 33.5 MB
    bf16_t* WT    = (bf16_t*)(base + off); off += (size_t)FOURH * KTOT * 2;              //  6.3 MB
    float*  hf    = (float* )(base + off); off += (size_t)N_BATCH * H_DIM * 4;           //  1.0 MB
    bf16_t* hb    = (bf16_t*)(base + off); off += (size_t)N_BATCH * H_DIM * 2;
    bf16_t* attnb = (bf16_t*)(base + off); off += (size_t)N_BATCH * H_DIM * 2;
    float*  cst   = (float* )(base + off); off += (size_t)N_BATCH * H_DIM * 4;
    float*  aout  = (float* )(base + off); off += (size_t)N_BATCH * FOURH * 4;           //  4.2 MB
    (void)off; (void)ws_size; (void)in_sizes; (void)n_in; (void)out_size;

    // prep: cast x to bf16, build transposed concatenated weights, init h0/c0
    k_cast_bf16<<<(N_BATCH * T_STEPS * D_IN) / 256, 256, 0, stream>>>(x, xbf);
    k_build_wt <<<(FOURH * KTOT) / 256,            256, 0, stream>>>(Wx, Wh, Wattn, WT);
    k_init     <<<(N_BATCH * H_DIM) / 256,         256, 0, stream>>>(A, hf, hb, cst);

    // recurrence: 64 sequential steps of attention -> fused WMMA GEMM -> gates
    for (int t = 0; t < T_STEPS; ++t) {
        k_attn<<<N_BATCH, 256, 0, stream>>>(A, hf, attnb);
        k_gemm<<<dim3(4, 16), 256, 0, stream>>>(xbf, hb, attnb, WT, aout, t);
        k_gate<<<(N_BATCH * H_DIM) / 256, 256, 0, stream>>>(aout, b, cst, hf, hb, out, t);
    }
}
